// BaseAttn_32658931318995
// MI455X (gfx1250) — matrix operand order repaired and checked by reading, not compiled, not run on hardware
//
#include <hip/hip_runtime.h>

typedef __attribute__((ext_vector_type(16))) _Float16 v16h;
typedef __attribute__((ext_vector_type(8)))  float    v8f;

#define B_SZ   32768
#define H_SZ   16
#define D_SZ   128
#define RSTR   132   // padded row stride (floats) for the per-wave tile buffer
#define WAVES  2
#define NBLK   2048  // NBLK*WAVES waves -> 8 batches per wave

// ---- fast transcendentals (single trans-op where available) ----
__device__ __forceinline__ float fast_exp2(float x) {
#if __has_builtin(__builtin_amdgcn_exp2f)
    return __builtin_amdgcn_exp2f(x);
#else
    return exp2f(x);
#endif
}
__device__ __forceinline__ float fast_rcp(float x) {
#if __has_builtin(__builtin_amdgcn_rcpf)
    return __builtin_amdgcn_rcpf(x);
#else
    return 1.0f / x;
#endif
}
__device__ __forceinline__ float fast_tanh(float x) {
    float xc = fminf(fmaxf(x, -16.0f), 16.0f);          // avoid inf/inf
    float e  = fast_exp2(2.88539008f * xc);             // exp(2x)
    return 1.0f - 2.0f * fast_rcp(e + 1.0f);
}

// ---- CDNA5 async global->LDS staging of a 16x128 f32 tile ----
__device__ __forceinline__ void wait_async0() {
    asm volatile("s_wait_asynccnt 0" ::: "memory");
}
__device__ __forceinline__ void wait_ds0() {
    asm volatile("s_wait_dscnt 0" ::: "memory");
}
__device__ __forceinline__ void stage_tile_async(float* buf, const float* gsrc, int lane) {
    #pragma unroll
    for (int it = 0; it < 16; ++it) {
        const int idx  = lane + 32 * it;      // float4 index within 16x128 tile
        const int row  = idx >> 5;            // 32 float4 per row
        const int col4 = idx & 31;
        uint32_t ldsb = (uint32_t)(size_t)(const void*)&buf[row * RSTR + col4 * 4];
        const float4* g = (const float4*)gsrc + idx;
        asm volatile("global_load_async_to_lds_b128 %0, %1, off"
                     :: "v"(ldsb), "v"(g) : "memory");
    }
    wait_async0();
}

__global__ __launch_bounds__(WAVES * 32)
void attn_combine_hops_kernel(const float* __restrict__ r1,
                              const float* __restrict__ r2,
                              const float* __restrict__ W,
                              float* __restrict__ out) {
    // W pre-swizzled into exact per-lane WMMA B fragments:
    // [chunk c][tile t][lane][16 halves] -> 32 KB, conflict-free 32B reads
    __shared__ _Float16 Wfrag[4 * 8 * 32 * 16];
    // one shared staging buffer per wave (r1 first, then r2)
    __shared__ float    buf_all[WAVES][16 * RSTR];

    const int tid  = threadIdx.x;
    const int w    = tid >> 5;
    const int lane = tid & 31;
    const int lo   = lane & 15;
    const int hi   = lane >> 4;
    float* buf = buf_all[w];

    // ---- stage W (f32 -> f16), swizzled to B-fragment layout ----
    // element j of lane ln in (c,t): N-col = t*16 + (ln&15), K = c*32 + 16*(ln>>4) + j
    for (int idx = tid; idx < 4 * 8 * 32 * 16; idx += WAVES * 32) {
        const int j  = idx & 15;
        const int ln = (idx >> 4) & 31;
        const int tt = (idx >> 9) & 7;
        const int cc = idx >> 12;
        Wfrag[idx] = (_Float16)W[(tt * 16 + (ln & 15)) * D_SZ +
                                 cc * 32 + 16 * (ln >> 4) + j];
    }
    __syncthreads();

    const int wavesTotal = gridDim.x * WAVES;
    for (int b = blockIdx.x * WAVES + w; b < B_SZ; b += wavesTotal) {
        const float* r1b = r1 + (size_t)b * (H_SZ * D_SZ);
        const float* r2b = r2 + (size_t)b * (H_SZ * D_SZ);

        // ---- stage r1 tile (async DMA to LDS), pull fragments to registers ----
        wait_ds0();                          // prior iter's LDS reads done
        stage_tile_async(buf, r1b, lane);
        // C/D-layout fragments: lane holds r1[m][e], m = v + 8*hi, e = t*16 + lo
        float r1f[8][8];
        #pragma unroll
        for (int t = 0; t < 8; ++t)
            #pragma unroll
            for (int v = 0; v < 8; ++v)
                r1f[t][v] = buf[(v + 8 * hi) * RSTR + t * 16 + lo];

        // ---- stage r2 tile into the same buffer ----
        wait_ds0();                          // r1 fragment reads done
        stage_tile_async(buf, r2b, lane);

        // ---- P = r2_tile(16x128) @ W^T : 8 e-tiles x 4 K-chunks ----
        v8f acc[8] = {};
        const int base = 16 * hi;
        #pragma unroll
        for (int c = 0; c < 4; ++c) {
            // A fragment: row m = lo, K = c*32 + 16*hi + j (matches Wfrag)
            const float* arow = &buf[lo * RSTR + c * 32 + base];
            v16h a;
            #pragma unroll
            for (int j = 0; j < 8; ++j) a[j]     = (_Float16)arow[j];
            #pragma unroll
            for (int j = 0; j < 8; ++j) a[8 + j] = (_Float16)arow[8 + j];

            #pragma unroll
            for (int t = 0; t < 8; ++t) {
                const v16h bb = *(const v16h*)&Wfrag[(((c << 3) + t) * 32 + lane) * 16];
                acc[t] = __builtin_amdgcn_wmma_f32_16x16x32_f16(
                    false, a, false, bb, (short)0, acc[t], false, false);
            }
        }

        // ---- scores[m] = sum_e r1[m][e] * P[m][e] ----
        float dv[8];
        #pragma unroll
        for (int v = 0; v < 8; ++v) {
            float s = 0.f;
            #pragma unroll
            for (int t = 0; t < 8; ++t) s += acc[t][v] * r1f[t][v];
            #pragma unroll
            for (int m = 1; m < 16; m <<= 1) s += __shfl_xor(s, m, 16);
            dv[v] = s;   // row v (low half) / row v+8 (high half)
        }
        float scl[8], sch[8];   // scores rows 0..7 / 8..15, in every lane
        #pragma unroll
        for (int v = 0; v < 8; ++v) {
            float other = __shfl_xor(dv[v], 16, 32);
            scl[v] = hi ? other : dv[v];
            sch[v] = hi ? dv[v] : other;
        }

        // ---- tanh + softmax over 16 hops (in-lane, trans-op based) ----
        float mx = -1e30f;
        #pragma unroll
        for (int h = 0; h < 8; ++h) {
            scl[h] = fast_tanh(scl[h]);
            sch[h] = fast_tanh(sch[h]);
            mx = fmaxf(mx, fmaxf(scl[h], sch[h]));
        }
        float sum = 0.f;
        #pragma unroll
        for (int h = 0; h < 8; ++h) {
            scl[h] = fast_exp2(1.44269504f * (scl[h] - mx));
            sch[h] = fast_exp2(1.44269504f * (sch[h] - mx));
            sum += scl[h] + sch[h];
        }
        const float inv = fast_rcp(sum);
        #pragma unroll
        for (int h = 0; h < 8; ++h) { scl[h] *= inv; sch[h] *= inv; }

        // ---- adist output ----
        float myA = 0.f;
        #pragma unroll
        for (int h = 0; h < 8; ++h) {
            myA = (lo == h)     ? scl[h] : myA;
            myA = (lo == h + 8) ? sch[h] : myA;
        }
        if (hi == 0)
            out[(size_t)2 * B_SZ * D_SZ + (size_t)b * H_SZ + lo] = myA;

        // ---- r1c from registered fragments ----
        #pragma unroll
        for (int t = 0; t < 8; ++t) {
            float cacc = 0.f;
            #pragma unroll
            for (int v = 0; v < 8; ++v)
                cacc += (hi ? sch[v] : scl[v]) * r1f[t][v];
            cacc += __shfl_xor(cacc, 16, 32);   // add other half's 8 hops
            if (hi == 0)
                out[(size_t)b * D_SZ + t * 16 + lo] = cacc;
        }

        // ---- r2c from LDS-resident r2 ----
        #pragma unroll
        for (int k = 0; k < 4; ++k) {
            const int d = lane + 32 * k;
            float cacc = 0.f;
            #pragma unroll
            for (int h = 0; h < 8; ++h) {
                cacc += scl[h] * buf[h * RSTR + d];
                cacc += sch[h] * buf[(h + 8) * RSTR + d];
            }
            out[(size_t)B_SZ * D_SZ + (size_t)b * D_SZ + d] = cacc;
        }
    }
}

extern "C" void kernel_launch(void* const* d_in, const int* in_sizes, int n_in,
                              void* d_out, int out_size, void* d_ws, size_t ws_size,
                              hipStream_t stream) {
    (void)in_sizes; (void)n_in; (void)d_ws; (void)ws_size; (void)out_size;
    const float* r1 = (const float*)d_in[0];
    const float* r2 = (const float*)d_in[1];
    const float* W  = (const float*)d_in[2];
    float* out = (float*)d_out;
    attn_combine_hops_kernel<<<NBLK, WAVES * 32, 0, stream>>>(r1, r2, W, out);
}
